// MWKPN_88347477279304
// MI455X (gfx1250) — matrix operand order repaired and checked by reading, not compiled, not run on hardware
//
#include <hip/hip_runtime.h>
#include <hip/hip_bf16.h>

typedef __attribute__((ext_vector_type(16))) _Float16 v16h;
typedef __attribute__((ext_vector_type(8)))  _Float16 v8h;
typedef __attribute__((ext_vector_type(8)))  float    v8f;

static __device__ __forceinline__ float lrelu(float v) {
  return v >= 0.f ? v : 0.3f * v;
}

// ---------------- elementwise kernels ----------------

// out[b,y,x,n] = est[b,n,y,x]   (2,8,128,128,1) -> (2,128,128,8)
__global__ void k_transpose_in(const float* __restrict__ in, float* __restrict__ out) {
  int idx = blockIdx.x * blockDim.x + threadIdx.x;
  if (idx >= 2 * 128 * 128 * 8) return;
  int n = idx & 7; int t = idx >> 3;
  int x = t & 127; t >>= 7;
  int y = t & 127; int b = t >> 7;
  out[idx] = in[((size_t)((b * 8 + n) * 128 + y) * 128) + x];
}

// in: (2,H,W,C) -> out: (2,H/2,W/2,4C)
__global__ void k_dwt(const float* __restrict__ in, float* __restrict__ out,
                      int H, int W, int C) {
  int h2 = H >> 1, w2 = W >> 1;
  int total = 2 * h2 * w2 * C;
  int idx = blockIdx.x * blockDim.x + threadIdx.x;
  if (idx >= total) return;
  int c = idx % C; int t = idx / C;
  int j = t % w2; t /= w2;
  int i = t % h2; int b = t / h2;
  size_t base = ((size_t)(b * H + 2 * i) * W + 2 * j) * C + c;
  float s1 = in[base] * 0.5f;                       // even row, even col
  float s3 = in[base + C] * 0.5f;                   // even row, odd col
  float s2 = in[base + (size_t)W * C] * 0.5f;       // odd row, even col
  float s4 = in[base + (size_t)W * C + C] * 0.5f;   // odd row, odd col
  size_t ob = ((size_t)(b * h2 + i) * w2 + j) * (4 * C) + c;
  out[ob]           =  s1 + s2 + s3 + s4;
  out[ob + C]       = -s1 - s2 + s3 + s4;
  out[ob + 2 * C]   = -s1 + s2 - s3 + s4;
  out[ob + 3 * C]   =  s1 - s2 - s3 + s4;
}

// in: (2,h,w,4C) -> out: (2,2h,2w,C)
__global__ void k_idwt(const float* __restrict__ in, float* __restrict__ out,
                       int h, int w, int C) {
  int total = 2 * h * w * C;
  int idx = blockIdx.x * blockDim.x + threadIdx.x;
  if (idx >= total) return;
  int c = idx % C; int t = idx / C;
  int j = t % w; t /= w;
  int i = t % h; int b = t / h;
  size_t ib = ((size_t)(b * h + i) * w + j) * (4 * C);
  float a  = in[ib + c] * 0.5f;
  float bb = in[ib + C + c] * 0.5f;
  float cc = in[ib + 2 * C + c] * 0.5f;
  float dd = in[ib + 3 * C + c] * 0.5f;
  float ee = a - bb - cc + dd;
  float oe = a - bb + cc - dd;
  float eo = a + bb - cc - dd;
  float oo = a + bb + cc + dd;
  int W2 = 2 * w;
  size_t ob = ((size_t)(b * 2 * h + 2 * i) * W2 + 2 * j) * C + c;
  out[ob]                       = ee;  // (2i, 2j)
  out[ob + C]                   = eo;  // (2i, 2j+1)
  out[ob + (size_t)W2 * C]      = oe;  // (2i+1, 2j)
  out[ob + (size_t)W2 * C + C]  = oo;  // (2i+1, 2j+1)
}

// channel concat: out[m, 0:C1)=a, [C1:C1+C2)=b ; m over (batch,y,x)
__global__ void k_concat(const float* __restrict__ a, const float* __restrict__ b,
                         float* __restrict__ out, int M, int C1, int C2) {
  int C = C1 + C2;
  int total = M * C;
  int idx = blockIdx.x * blockDim.x + threadIdx.x;
  if (idx >= total) return;
  int c = idx % C; int m = idx / C;
  out[idx] = (c < C1) ? a[(size_t)m * C1 + c] : b[(size_t)m * C2 + (c - C1)];
}

// weight repack: HWIO f32 (flat [tap*Cin+ci][co]) -> f16 [co][tap*Cpad+ci],
// zero-filled for ci in [Cin, Cpad)  (so conv tail chunks need no guards)
__global__ void k_wtrans(const float* __restrict__ w, _Float16* __restrict__ wT,
                         int Cin, int Cpad, int Cout) {
  int total = 9 * Cpad * Cout;
  int idx = blockIdx.x * blockDim.x + threadIdx.x;
  if (idx >= total) return;
  int co = idx % Cout;
  int kk = idx / Cout;            // 0 .. 9*Cpad-1
  int tap = kk / Cpad;
  int ci  = kk - tap * Cpad;
  _Float16 v = (_Float16)0.f;
  if (ci < Cin) v = (_Float16)w[(size_t)(tap * Cin + ci) * Cout + co];
  wT[(size_t)co * (9 * Cpad) + kk] = v;
}

// ---------------- WMMA implicit-GEMM conv3x3 (SAME) + bias + leakyReLU ----
// M = 2*H*W (NHWC rows), N = Cout, K padded per tap to Cpad (mult of 32).
// 256 threads = 8 waves; block tile 128(M)x64(N); wave tile 32x32 = 4 WMMAs.
// Branchless loaders: spatial pad via clamp+zero-scale, k-tail via wT zeros.

#define LSTR 40   // halves per LDS row (80B: 16B-aligned, bank-spread)

__global__ __launch_bounds__(256) void k_conv3x3_wmma(
    const float* __restrict__ in, const _Float16* __restrict__ wT,
    const float* __restrict__ bias, float* __restrict__ out,
    int H, int W, int Cin, int Cout)
{
  __shared__ __align__(16) _Float16 As[128 * LSTR];
  __shared__ __align__(16) _Float16 Bs[64 * LSTR];   // stored [n][k]

  const int Cpad = ((Cin + 31) >> 5) << 5;
  const int K9   = 9 * Cpad;
  const int HW   = H * W;
  const int tid  = threadIdx.x;
  const int lane = tid & 31;
  const int wv   = tid >> 5;          // 0..7
  const int wm   = (wv & 3) * 32;     // wave M offset (2 tiles of 16)
  const int wn   = (wv >> 2) * 32;    // wave N offset (2 tiles of 16)

  const int m0  = blockIdx.y * 128;
  const int co0 = blockIdx.x * 64;

  // A loader: row = tid>>1 (0..127), 16 consecutive k per thread
  const int arow = tid >> 1;
  const int alk0 = (tid & 1) * 16;
  const int gm = m0 + arow;
  const int lb = gm / HW;
  const int lr = gm - lb * HW;
  const int ly = lr / W;
  const int lx = lr - ly * W;

  // B loader: row = tid>>2 (0..63), 8 consecutive k per thread
  const int brow = tid >> 2;
  const int blk0 = (tid & 3) * 8;
  const int lco  = min(co0 + brow, Cout - 1);   // clamp: padded cols never stored
  const _Float16* wrow = wT + (size_t)lco * K9;

  // fragment coordinates
  const int mr = lane & 15;
  const int hf = lane >> 4;

  const int steps = K9 >> 5;

  v8f acc00 = {}, acc01 = {}, acc10 = {}, acc11 = {};

  float va[16];     // prefetched activation chunk (f32)
  float ascale;     // 0/1 spatial-pad mask for the prefetched chunk
  v8h   vb;         // prefetched weight chunk (f16)

  // next-chunk counters (no divisions anywhere in the loop)
  int dy_n = -1, dx_n = -1, c0_n = 0, kg_n = 0;
  auto do_load = [&]() {
    int yy = ly + dy_n, xx = lx + dx_n;
    int yc = min(max(yy, 0), H - 1);
    int xc = min(max(xx, 0), W - 1);
    ascale = (yy == yc && xx == xc) ? 1.f : 0.f;
    const float* src = in + ((size_t)(lb * H + yc) * W + xc) * Cin + c0_n + alk0;
    const float4* f4 = (const float4*)src;       // 16B-aligned (Cin % 8 == 0)
    #pragma unroll
    for (int q = 0; q < 4; ++q) {
      float4 f = f4[q];
      va[4 * q + 0] = f.x; va[4 * q + 1] = f.y;
      va[4 * q + 2] = f.z; va[4 * q + 3] = f.w;
    }
    vb = *(const v8h*)(wrow + kg_n + blk0);      // 16B-aligned, zero-padded tail
    kg_n += 32;
    c0_n += 32;
    if (c0_n >= Cpad) {
      c0_n = 0;
      if (++dx_n > 1) { dx_n = -1; ++dy_n; }
    }
  };

  auto fragA = [&](int r0) -> v16h {
    const _Float16* rp = &As[r0 * LSTR];
    v8h lo = *(const v8h*)&rp[hf * 8];
    v8h hi = *(const v8h*)&rp[16 + hf * 8];
    return __builtin_shufflevector(lo, hi,
        0,1,2,3,4,5,6,7,8,9,10,11,12,13,14,15);
  };
  auto fragB = [&](int r0) -> v16h {
    const _Float16* rp = &Bs[r0 * LSTR];
    v8h lo = *(const v8h*)&rp[hf * 8];
    v8h hi = *(const v8h*)&rp[16 + hf * 8];
    return __builtin_shufflevector(lo, hi,
        0,1,2,3,4,5,6,7,8,9,10,11,12,13,14,15);
  };

  do_load();

  for (int s = 0; s < steps; ++s) {
    __syncthreads();                        // prior LDS reads complete
    {
      v8h ta0, ta1;
      #pragma unroll
      for (int u = 0; u < 8; ++u) ta0[u] = (_Float16)(va[u] * ascale);
      #pragma unroll
      for (int u = 0; u < 8; ++u) ta1[u] = (_Float16)(va[8 + u] * ascale);
      *(v8h*)&As[arow * LSTR + alk0]     = ta0;
      *(v8h*)&As[arow * LSTR + alk0 + 8] = ta1;
      *(v8h*)&Bs[brow * LSTR + blk0]     = vb;
    }
    __syncthreads();                        // tile visible

    if (s + 1 < steps) do_load();           // prefetch (overlaps WMMAs)

    v16h af0 = fragA(wm + mr);
    v16h af1 = fragA(wm + 16 + mr);
    v16h bf0 = fragB(wn + mr);
    v16h bf1 = fragB(wn + 16 + mr);

    acc00 = __builtin_amdgcn_wmma_f32_16x16x32_f16(
        false, af0, false, bf0, (short)0, acc00, false, false);
    acc01 = __builtin_amdgcn_wmma_f32_16x16x32_f16(
        false, af0, false, bf1, (short)0, acc01, false, false);
    acc10 = __builtin_amdgcn_wmma_f32_16x16x32_f16(
        false, af1, false, bf0, (short)0, acc10, false, false);
    acc11 = __builtin_amdgcn_wmma_f32_16x16x32_f16(
        false, af1, false, bf1, (short)0, acc11, false, false);
  }

  // epilogue: C/D layout: n = lane&15, VGPR r -> m = r + 8*(lane>=16)
  const int coA = co0 + wn + mr;
  const int coB = coA + 16;
  const int mbA = m0 + wm + hf * 8;
  const int mbB = mbA + 16;
  if (coA < Cout) {
    float bv = bias[coA];
    #pragma unroll
    for (int r = 0; r < 8; ++r)
      out[(size_t)(mbA + r) * Cout + coA] = lrelu(acc00[r] + bv);
    #pragma unroll
    for (int r = 0; r < 8; ++r)
      out[(size_t)(mbB + r) * Cout + coA] = lrelu(acc10[r] + bv);
  }
  if (coB < Cout) {
    float bv = bias[coB];
    #pragma unroll
    for (int r = 0; r < 8; ++r)
      out[(size_t)(mbA + r) * Cout + coB] = lrelu(acc01[r] + bv);
    #pragma unroll
    for (int r = 0; r < 8; ++r)
      out[(size_t)(mbB + r) * Cout + coB] = lrelu(acc11[r] + bv);
  }
}

// ---------------- final 5x5 kernel-prediction apply ----------------
// core flat (2,128,128,200); reference reshape identity:
//   core[b,n,y,x,0,t] = core_flat[b*3276800 + n*409600 + y*3200 + x*25 + t]
__global__ void k_kpn_apply(const float* __restrict__ core, const float* __restrict__ data,
                            float* __restrict__ pred_img, float* __restrict__ pred_img_i) {
  int idx = blockIdx.x * blockDim.x + threadIdx.x;
  if (idx >= 2 * 128 * 128) return;
  int x = idx & 127; int t = idx >> 7;
  int y = t & 127; int b = t >> 7;
  float mean = 0.f;
  for (int n = 0; n < 8; ++n) {
    size_t cb = (size_t)b * 3276800 + (size_t)n * 409600 + (size_t)y * 3200 + (size_t)x * 25;
    const float* dbase = data + (size_t)(b * 8 + n) * 128 * 128;
    float s = 0.f;
    #pragma unroll
    for (int i = 0; i < 5; ++i) {
      int yy = y + i - 2;
      #pragma unroll
      for (int j = 0; j < 5; ++j) {
        int xx = x + j - 2;
        float dv = (yy >= 0 && yy < 128 && xx >= 0 && xx < 128) ? dbase[yy * 128 + xx] : 0.f;
        s += core[cb + i * 5 + j] * dv;
      }
    }
    pred_img_i[((size_t)(b * 8 + n) * 128 + y) * 128 + x] = s;
    mean += s;
  }
  pred_img[((size_t)(b * 128) + y) * 128 + x] = mean * 0.125f;
}

// ---------------- host orchestration ----------------

extern "C" void kernel_launch(void* const* d_in, const int* in_sizes, int n_in,
                              void* d_out, int out_size, void* d_ws, size_t ws_size,
                              hipStream_t stream) {
  (void)in_sizes; (void)n_in; (void)out_size; (void)ws_size;
  const float* est  = (const float*)d_in[0];
  const float* data = (const float*)d_in[1];
  const float* Wt[8][3]; const float* Bi[8][3];
  int p = 2;
  for (int b = 0; b < 8; ++b)
    for (int i = 0; i < 3; ++i) { Wt[b][i] = (const float*)d_in[p++]; Bi[b][i] = (const float*)d_in[p++]; }

  float* ws = (float*)d_ws;
  // bump-allocated early pool (dead before oA/oB are written)
  float* x    = ws + 0;          // 262144
  float* d1   = ws + 262144;     // 262144
  float* t1a  = ws + 524288;     // 524288
  float* t1b  = ws + 1048576;    // 524288
  float* d2   = ws + 1572864;    // 524288
  float* t2a  = ws + 2097152;    // 262144
  float* t2b  = ws + 2359296;    // 262144
  float* d3   = ws + 2621440;    // 262144
  float* t3a  = ws + 2883584;    // 98304
  float* t3b  = ws + 2981888;    // 98304
  float* d4   = ws + 3080192;    // 98304
  float* t4a  = ws + 3178496;    // 32768
  float* t4b  = ws + 3211264;    // 32768
  float* u4   = ws + 3244032;    // 32768
  float* cat5 = ws + 3276800;    // 131072
  float* t5a  = ws + 3407872;    // 98304
  float* t5b  = ws + 3506176;    // 98304
  float* u5   = ws + 3604480;    // 98304
  float* cat6 = ws + 3702784;    // 360448
  float* t6a  = ws + 4063232;    // 262144
  float* t6b  = ws + 4325376;    // 262144
  float* u6   = ws + 4587520;    // 262144
  float* cat7 = ws + 4849664;    // 786432
  float* t7a  = ws + 5636096;    // 524288
  float* t7b  = ws + 6160384;    // 524288  (early pool ends at 6684672)
  float* u7   = ws + 13107200;   // 524288  (above oB)
  float* oA   = ws + 0;          // 6553600 (reuses dead early pool)
  float* oB   = ws + 6553600;    // 6553600
  _Float16* hbase = (_Float16*)(ws + 13631488);   // f16 weight pool (~7.8M halves)

  static const int specs[8][3][2] = {
    {{32,64},{64,64},{64,64}},
    {{256,128},{128,128},{128,128}},
    {{512,192},{192,192},{192,192}},
    {{768,256},{256,256},{256,256}},
    {{256,192},{192,192},{192,192}},
    {{176,128},{128,128},{128,128}},
    {{96,64},{64,64},{64,64}},
    {{16,200},{200,200},{200,200}},
  };

  auto cdiv = [](int a, int b){ return (a + b - 1) / b; };

  // one-time (per launch) weight repack to f16 [co][tap*Cpad+ci], zero-padded
  _Float16* wTp[8][3];
  size_t hoff = 0;
  for (int b = 0; b < 8; ++b)
    for (int i = 0; i < 3; ++i) {
      int Cin = specs[b][i][0], Cout = specs[b][i][1];
      int Cpad = ((Cin + 31) / 32) * 32;
      int K9 = 9 * Cpad;
      wTp[b][i] = hbase + hoff;
      hoff += (size_t)K9 * Cout;
      k_wtrans<<<dim3((unsigned)cdiv(K9 * Cout, 256)), 256, 0, stream>>>(
          Wt[b][i], wTp[b][i], Cin, Cpad, Cout);
    }

  auto conv = [&](const float* in, const _Float16* w, const float* bb, float* out,
                  int H, int W, int Cin, int Cout) {
    dim3 grid(cdiv(Cout, 64), (2 * H * W) / 128);
    k_conv3x3_wmma<<<grid, 256, 0, stream>>>(in, w, bb, out, H, W, Cin, Cout);
  };
  auto ewg = [&](int total){ return dim3((unsigned)((total + 255) / 256)); };

  k_transpose_in<<<ewg(2*128*128*8), 256, 0, stream>>>(est, x);

  k_dwt<<<ewg(2*64*64*8), 256, 0, stream>>>(x, d1, 128, 128, 8);
  conv(d1,  wTp[0][0], Bi[0][0], t1a, 64, 64, 32, 64);
  conv(t1a, wTp[0][1], Bi[0][1], t1b, 64, 64, 64, 64);
  conv(t1b, wTp[0][2], Bi[0][2], t1a, 64, 64, 64, 64);         // c1 = t1a

  k_dwt<<<ewg(2*32*32*64), 256, 0, stream>>>(t1a, d2, 64, 64, 64);
  conv(d2,  wTp[1][0], Bi[1][0], t2a, 32, 32, 256, 128);
  conv(t2a, wTp[1][1], Bi[1][1], t2b, 32, 32, 128, 128);
  conv(t2b, wTp[1][2], Bi[1][2], t2a, 32, 32, 128, 128);       // c2 = t2a

  k_dwt<<<ewg(2*16*16*128), 256, 0, stream>>>(t2a, d3, 32, 32, 128);
  conv(d3,  wTp[2][0], Bi[2][0], t3a, 16, 16, 512, 192);
  conv(t3a, wTp[2][1], Bi[2][1], t3b, 16, 16, 192, 192);
  conv(t3b, wTp[2][2], Bi[2][2], t3a, 16, 16, 192, 192);       // c3 = t3a

  k_dwt<<<ewg(2*8*8*192), 256, 0, stream>>>(t3a, d4, 16, 16, 192);
  conv(d4,  wTp[3][0], Bi[3][0], t4a, 8, 8, 768, 256);
  conv(t4a, wTp[3][1], Bi[3][1], t4b, 8, 8, 256, 256);
  conv(t4b, wTp[3][2], Bi[3][2], t4a, 8, 8, 256, 256);         // c4 = t4a

  k_idwt<<<ewg(2*8*8*64), 256, 0, stream>>>(t4a, u4, 8, 8, 64);
  k_concat<<<ewg(2*16*16*256), 256, 0, stream>>>(t3a, u4, cat5, 2*16*16, 192, 64);
  conv(cat5, wTp[4][0], Bi[4][0], t5a, 16, 16, 256, 192);
  conv(t5a,  wTp[4][1], Bi[4][1], t5b, 16, 16, 192, 192);
  conv(t5b,  wTp[4][2], Bi[4][2], t5a, 16, 16, 192, 192);      // c5 = t5a

  k_idwt<<<ewg(2*16*16*48), 256, 0, stream>>>(t5a, u5, 16, 16, 48);
  k_concat<<<ewg(2*32*32*176), 256, 0, stream>>>(t2a, u5, cat6, 2*32*32, 128, 48);
  conv(cat6, wTp[5][0], Bi[5][0], t6a, 32, 32, 176, 128);
  conv(t6a,  wTp[5][1], Bi[5][1], t6b, 32, 32, 128, 128);
  conv(t6b,  wTp[5][2], Bi[5][2], t6a, 32, 32, 128, 128);      // c6 = t6a

  k_idwt<<<ewg(2*32*32*32), 256, 0, stream>>>(t6a, u6, 32, 32, 32);
  k_concat<<<ewg(2*64*64*96), 256, 0, stream>>>(t1a, u6, cat7, 2*64*64, 64, 32);
  conv(cat7, wTp[6][0], Bi[6][0], t7a, 64, 64, 96, 64);
  conv(t7a,  wTp[6][1], Bi[6][1], t7b, 64, 64, 64, 64);
  conv(t7b,  wTp[6][2], Bi[6][2], t7a, 64, 64, 64, 64);        // c7 = t7a

  k_idwt<<<ewg(2*64*64*16), 256, 0, stream>>>(t7a, u7, 64, 64, 16);
  conv(u7, wTp[7][0], Bi[7][0], oA, 128, 128, 16, 200);
  conv(oA, wTp[7][1], Bi[7][1], oB, 128, 128, 200, 200);
  conv(oB, wTp[7][2], Bi[7][2], oA, 128, 128, 200, 200);       // core = oA

  float* pred = (float*)d_out;
  k_kpn_apply<<<ewg(2*128*128), 256, 0, stream>>>(oA, data, pred, pred + 32768);
}
